// Mutihead_Attention_18863496364320
// MI455X (gfx1250) — matrix operand order repaired and checked by reading, not compiled, not run on hardware
//
#include <hip/hip_runtime.h>
#include <math.h>

// ---------------------------------------------------------------------------
// Problem constants (from reference)
// ---------------------------------------------------------------------------
#define D_MODEL 512
#define SEQ     2048
#define NHB     16          // N_HEADS * B = 8 * 2
#define HD      64          // head dim (512 / 8)
#define ROWS    4096        // B * S

typedef __bf16 bf16_t;
typedef __attribute__((ext_vector_type(16))) __bf16 v16bf;
typedef __attribute__((ext_vector_type(8)))  float  v8f;
typedef unsigned int u32x4 __attribute__((ext_vector_type(4)));
typedef int          i32x4 __attribute__((ext_vector_type(4)));
typedef int          i32x8 __attribute__((ext_vector_type(8)));

// ---------------------------------------------------------------------------
// WMMA tile loaders (wave32, v_wmma_f32_16x16x32_bf16 operand layouts)
//
// A-matrix 16x32 (MxK), 16-bit: lane L holds row M = L&15.
//   VGPR0-3  : K = klo..klo+7        (klo = (L>>4)*8)
//   VGPR4-7  : K = 16+klo..16+klo+7
// Memory operand is row-major [M][K] (K contiguous), row stride `stride`.
// ---------------------------------------------------------------------------
__device__ __forceinline__ v16bf load_atile16x32(const bf16_t* base, int stride, int lane) {
    int r = lane & 15;
    int klo = (lane >> 4) * 8;
    const bf16_t* p = base + (size_t)r * stride + klo;
    union { uint4 u[2]; v16bf v; } t;
    t.u[0] = *(const uint4*)(p);        // K = klo .. klo+7
    t.u[1] = *(const uint4*)(p + 16);   // K = 16+klo .. 16+klo+7
    return t.v;
}

// B-matrix 32x16 (KxN), 16-bit: lane L holds column N = L&15.
//   elements 0..7 : K = kb + 0..7, elements 8..15 : K = 16+kb + 0..7   (kb = (L>>4)*8)
// Memory operand is stored [N][K] (K contiguous), row stride `stride`.
// Works for both global and LDS-derived generic pointers.
__device__ __forceinline__ v16bf load_btile32x16(const bf16_t* base, int stride, int lane) {
    int n = lane & 15;
    int kb = (lane >> 4) * 8;
    const bf16_t* p = base + (size_t)n * stride + kb;
    union { uint4 u[2]; v16bf v; } t;
    t.u[0] = *(const uint4*)(p);        // K = kb .. kb+7
    t.u[1] = *(const uint4*)(p + 16);   // K = 16+kb .. 16+kb+7
    return t.v;
}

// ---------------------------------------------------------------------------
// f32 -> bf16 conversion
// ---------------------------------------------------------------------------
__global__ void f32_to_bf16_kernel(const float* __restrict__ src,
                                   bf16_t* __restrict__ dst, int n) {
    int i = blockIdx.x * blockDim.x + threadIdx.x;
    if (i < n) dst[i] = (bf16_t)src[i];
}

// ---------------------------------------------------------------------------
// V [hb][k][d] (bf16) -> Vt [hb][d][k] (bf16): makes PV B-operand K-contiguous
// ---------------------------------------------------------------------------
__global__ void transpose_v_kernel(const bf16_t* __restrict__ V,
                                   bf16_t* __restrict__ Vt) {
    int i = blockIdx.x * blockDim.x + threadIdx.x;
    if (i >= NHB * HD * SEQ) return;
    int hb  = i >> 17;               // / (HD*SEQ)
    int rem = i & (HD * SEQ - 1);
    int d   = rem >> 11;             // / SEQ
    int k   = rem & (SEQ - 1);
    Vt[i] = V[(size_t)hb * SEQ * HD + (size_t)k * HD + d];
}

// ---------------------------------------------------------------------------
// Stage a 64-row x 512-col bf16 weight panel (row-major, K contiguous) into
// LDS. Preferred path: Tensor Data Mover (single DMA descriptor, tracked by
// TENSORcnt). Fallback: cooperative vector copy.
// ---------------------------------------------------------------------------
__device__ __forceinline__ void stage_w_panel(bf16_t* __restrict__ Wlds,
                                              const bf16_t* __restrict__ Wsrc) {
#if defined(__gfx1250__) && __has_builtin(__builtin_amdgcn_tensor_load_to_lds)
    if ((threadIdx.x >> 5) == 0) {  // one wave issues the DMA
        unsigned long long ga = (unsigned long long)(uintptr_t)Wsrc;
        unsigned lds_addr = (unsigned)(uintptr_t)Wlds;  // low 32 bits = LDS offset
        // D# group 0: count=1(valid), lds_addr, global_addr[56:0], type=2
        u32x4 g0;
        g0[0] = 0x1u;
        g0[1] = lds_addr;
        g0[2] = (unsigned)ga;
        g0[3] = ((unsigned)(ga >> 32) & 0x01FFFFFFu) | 0x80000000u;  // type=2
        // D# group 1: data_size=2B(enc 1); tensor 512x64; tile 512x64;
        //             dim0_stride=512 elements
        i32x8 g1;
        g1[0] = 0x00010000;                 // data_size=1 (2 bytes)
        g1[1] = (int)(512u << 16);          // tensor_dim0[15:0]=512
        g1[2] = (int)(64u << 16);           // tensor_dim1[15:0]=64
        g1[3] = (int)(512u << 16);          // tile_dim0=512
        g1[4] = 64;                         // tile_dim1=64, tile_dim2=0
        g1[5] = 512;                        // tensor_dim0_stride[31:0]=512
        g1[6] = 0;
        g1[7] = 0;
        i32x4 g2 = {0, 0, 0, 0};            // 2D: higher dims unused
        i32x4 g3 = {0, 0, 0, 0};
#if __clang_major__ >= 23
        i32x8 g4 = {0, 0, 0, 0, 0, 0, 0, 0};
        __builtin_amdgcn_tensor_load_to_lds(g0, g1, g2, g3, g4, 0);
#else
        __builtin_amdgcn_tensor_load_to_lds(g0, g1, g2, g3, 0);
#endif
        __builtin_amdgcn_s_wait_tensorcnt(0);
    }
#else
    // Fallback: cooperative 16-byte copies (128 threads, 64*512*2 bytes)
    for (int c = threadIdx.x; c < 64 * D_MODEL / 8; c += 128) {
        int row = c >> 6;          // 64 chunks of 8 elems per 512-elem row
        int chunk = c & 63;
        *(uint4*)&Wlds[row * D_MODEL + chunk * 8] =
            *(const uint4*)&Wsrc[(size_t)row * D_MODEL + chunk * 8];
    }
#endif
    __syncthreads();
}

// ---------------------------------------------------------------------------
// GEMM: C[ROWS,512] = A[ROWS,512](bf16) @ W[512,512]^T(bf16) + bias(f32)
// W is torch nn.Linear layout [out,in] so both operands are K-contiguous.
// Block = 128 threads (4 waves); each wave computes a 16x64 strip.
// The 64x512 W panel (shared by all 4 waves) is staged in LDS via TDM.
// ---------------------------------------------------------------------------
template <typename OUT_T>
__device__ __forceinline__ void gemm_rowmajor_bt(const bf16_t* __restrict__ A,
                                                 const bf16_t* __restrict__ W,
                                                 const float*  __restrict__ bias,
                                                 OUT_T* __restrict__ C) {
    __shared__ __align__(16) bf16_t Wlds[64 * D_MODEL];   // 64 KB

    const int lane = threadIdx.x & 31;
    const int wave = threadIdx.x >> 5;
    const int n0 = blockIdx.x * 64;
    const int m0 = blockIdx.y * 64 + wave * 16;

    stage_w_panel(Wlds, W + (size_t)n0 * D_MODEL);

    const bf16_t* Arow = A + (size_t)(m0 + (lane & 15)) * D_MODEL;

    v8f acc[4] = {{}, {}, {}, {}};
    for (int kb = 0; kb < D_MODEL; kb += 32) {
        v16bf a = load_atile16x32(A + (size_t)m0 * D_MODEL + kb, D_MODEL, lane);
        if (kb + 32 < D_MODEL)
            __builtin_prefetch(Arow + kb + 32, 0, 1);     // global_prefetch_b8
#pragma unroll
        for (int s = 0; s < 4; ++s) {
            v16bf b = load_btile32x16(Wlds + (size_t)(s * 16) * D_MODEL + kb,
                                      D_MODEL, lane);      // ds_load_b128 x2
            acc[s] = __builtin_amdgcn_wmma_f32_16x16x32_bf16(
                false, a, false, b, (short)0, acc[s], false, false);
        }
    }

    const int r = lane & 15, h = lane >> 4;
#pragma unroll
    for (int s = 0; s < 4; ++s) {
        int col = n0 + s * 16 + r;
        float bv = bias[col];
#pragma unroll
        for (int i = 0; i < 8; ++i) {
            int row = m0 + i + h * 8;   // C layout: lanes 16-31 carry M = 8+i
            C[(size_t)row * D_MODEL + col] = (OUT_T)(acc[s][i] + bv);
        }
    }
}

__global__ __launch_bounds__(128) void gemm_qkv_kernel(
    const bf16_t* __restrict__ xb, const bf16_t* __restrict__ yb,
    const bf16_t* __restrict__ wq, const bf16_t* __restrict__ wk,
    const bf16_t* __restrict__ wv,
    const float* __restrict__ bq, const float* __restrict__ bk,
    const float* __restrict__ bv,
    bf16_t* __restrict__ Q, bf16_t* __restrict__ K, bf16_t* __restrict__ V) {
    const bf16_t* A; const bf16_t* W; const float* b; bf16_t* C;
    if (blockIdx.z == 0)      { A = xb; W = wq; b = bq; C = Q; }
    else if (blockIdx.z == 1) { A = yb; W = wk; b = bk; C = K; }
    else                      { A = yb; W = wv; b = bv; C = V; }
    gemm_rowmajor_bt<bf16_t>(A, W, b, C);
}

__global__ __launch_bounds__(128) void gemm_out_kernel(
    const bf16_t* __restrict__ A, const bf16_t* __restrict__ W,
    const float* __restrict__ bias, float* __restrict__ C) {
    gemm_rowmajor_bt<float>(A, W, bias, C);
}

// ---------------------------------------------------------------------------
// Attention: one workgroup per (hb, 16-query tile).
//  Phase 1: scores[16][2048] = (Q_tile @ K^T) * (1/sqrt(512))  -> LDS (f32)
//  Phase 2: row mean/max; mask (s<=mean -> 0 prob); exp(s-max) in place; denom
//  Phase 3: O = (P @ V) / denom via WMMA with A-tiles rebuilt from LDS
// ---------------------------------------------------------------------------
__global__ __launch_bounds__(128) void attn_kernel(
    const bf16_t* __restrict__ Qb, const bf16_t* __restrict__ Kb,
    const bf16_t* __restrict__ Vt, bf16_t* __restrict__ O) {
    __shared__ float sc[16][SEQ];        // 128 KB score row-block
    __shared__ float red[16][8];
    __shared__ float redm[16][8];
    __shared__ float rowmean[16];
    __shared__ float rowmax_s[16];
    __shared__ float rowden[16];
    __shared__ float opart[4][16][64];   // 16 KB cross-wave partials

    const int tid  = threadIdx.x;
    const int lane = tid & 31;
    const int wave = tid >> 5;
    const int hb = blockIdx.y;
    const int q0 = blockIdx.x * 16;

    const bf16_t* Qh  = Qb + (size_t)hb * SEQ * HD;
    const bf16_t* Kh  = Kb + (size_t)hb * SEQ * HD;
    const bf16_t* Vth = Vt + (size_t)hb * HD * SEQ;

    const float norm = 0.04419417382415922f;  // 1/sqrt(512)

    // ---- Phase 1: scores via WMMA; wave w covers keys [w*512, w*512+512) ----
    {
        v16bf a0 = load_atile16x32(Qh + (size_t)q0 * HD + 0,  HD, lane);
        v16bf a1 = load_atile16x32(Qh + (size_t)q0 * HD + 32, HD, lane);
        const int r = lane & 15, h = lane >> 4;
        for (int t = 0; t < 32; ++t) {
            int kb = wave * 512 + t * 16;
            v16bf b0 = load_btile32x16(Kh + (size_t)kb * HD + 0,  HD, lane);
            v16bf b1 = load_btile32x16(Kh + (size_t)kb * HD + 32, HD, lane);
            v8f c = {};
            c = __builtin_amdgcn_wmma_f32_16x16x32_bf16(false, a0, false, b0,
                                                        (short)0, c, false, false);
            c = __builtin_amdgcn_wmma_f32_16x16x32_bf16(false, a1, false, b1,
                                                        (short)0, c, false, false);
#pragma unroll
            for (int i = 0; i < 8; ++i)
                sc[i + h * 8][kb + r] = c[i] * norm;
        }
    }
    __syncthreads();

    // ---- Phase 2: row stats + mask + exp (8 threads per row) ----
    {
        const int r = tid >> 3;
        const int k = tid & 7;
        float mx = -3.4e38f, sum = 0.f;
        for (int c = k; c < SEQ; c += 8) {
            float v = sc[r][c];
            sum += v;
            mx = fmaxf(mx, v);
        }
        red[r][k]  = sum;
        redm[r][k] = mx;
        __syncthreads();
        if (tid < 16) {
            float su = 0.f, m2 = -3.4e38f;
#pragma unroll
            for (int j = 0; j < 8; ++j) { su += red[tid][j]; m2 = fmaxf(m2, redm[tid][j]); }
            rowmean[tid]  = su * (1.0f / SEQ);
            rowmax_s[tid] = m2;
        }
        __syncthreads();
        const float mean = rowmean[r], m2 = rowmax_s[r];
        float dsum = 0.f;
        for (int c = k; c < SEQ; c += 8) {
            float v = sc[r][c];
            float p = (v > mean) ? __expf(v - m2) : 0.f;  // s<=mean masked to -inf
            sc[r][c] = p;
            dsum += p;
        }
        red[r][k] = dsum;
        __syncthreads();
        if (tid < 16) {
            float su = 0.f;
#pragma unroll
            for (int j = 0; j < 8; ++j) su += red[tid][j];
            rowden[tid] = su;
        }
        __syncthreads();
    }

    // ---- Phase 3: O partial = P @ V over this wave's key range ----
    {
        const int r = lane & 15, h = lane >> 4;
        const int klo = h * 8;
        v8f acc[4] = {{}, {}, {}, {}};
        for (int t = 0; t < 16; ++t) {
            int kb = wave * 512 + t * 32;
            v16bf a;
#pragma unroll
            for (int e = 0; e < 8; ++e)
                a[e] = (bf16_t)sc[r][kb + klo + e];
#pragma unroll
            for (int e = 0; e < 8; ++e)
                a[8 + e] = (bf16_t)sc[r][kb + 16 + klo + e];
#pragma unroll
            for (int s = 0; s < 4; ++s) {
                v16bf b = load_btile32x16(Vth + (size_t)(s * 16) * SEQ + kb,
                                          SEQ, lane);
                acc[s] = __builtin_amdgcn_wmma_f32_16x16x32_bf16(
                    false, a, false, b, (short)0, acc[s], false, false);
            }
        }
#pragma unroll
        for (int s = 0; s < 4; ++s)
#pragma unroll
            for (int i = 0; i < 8; ++i)
                opart[wave][i + h * 8][s * 16 + r] = acc[s][i];
    }
    __syncthreads();

    // ---- Reduce 4 wave partials, divide by denom, store bf16 ----
    for (int idx = tid; idx < 16 * 64; idx += 128) {
        int m = idx >> 6, d = idx & 63;
        float o = (opart[0][m][d] + opart[1][m][d] +
                   opart[2][m][d] + opart[3][m][d]) / rowden[m];
        O[(size_t)hb * SEQ * HD + (size_t)(q0 + m) * HD + d] = (bf16_t)o;
    }
}

// ---------------------------------------------------------------------------
// Launch
// ---------------------------------------------------------------------------
extern "C" void kernel_launch(void* const* d_in, const int* in_sizes, int n_in,
                              void* d_out, int out_size, void* d_ws, size_t ws_size,
                              hipStream_t stream) {
    (void)in_sizes; (void)n_in; (void)out_size; (void)ws_size;
    const float* x   = (const float*)d_in[0];
    const float* y   = (const float*)d_in[1];
    const float* q_w = (const float*)d_in[2];
    const float* q_b = (const float*)d_in[3];
    const float* k_w = (const float*)d_in[4];
    const float* k_b = (const float*)d_in[5];
    const float* v_w = (const float*)d_in[6];
    const float* v_b = (const float*)d_in[7];
    const float* o_w = (const float*)d_in[8];
    const float* o_b = (const float*)d_in[9];
    float* out = (float*)d_out;

    char* ws = (char*)d_ws;
    size_t off = 0;
    auto alloc = [&](size_t bytes) -> void* {
        void* p = ws + off;
        off = (off + bytes + 255) & ~(size_t)255;
        return p;
    };
    bf16_t* xb   = (bf16_t*)alloc((size_t)ROWS * D_MODEL * 2);
    bf16_t* yb   = (bf16_t*)alloc((size_t)ROWS * D_MODEL * 2);
    bf16_t* wqb  = (bf16_t*)alloc((size_t)D_MODEL * D_MODEL * 2);
    bf16_t* wkb  = (bf16_t*)alloc((size_t)D_MODEL * D_MODEL * 2);
    bf16_t* wvb  = (bf16_t*)alloc((size_t)D_MODEL * D_MODEL * 2);
    bf16_t* wob  = (bf16_t*)alloc((size_t)D_MODEL * D_MODEL * 2);
    bf16_t* Qb   = (bf16_t*)alloc((size_t)ROWS * D_MODEL * 2);
    bf16_t* Kb   = (bf16_t*)alloc((size_t)ROWS * D_MODEL * 2);
    bf16_t* Vb   = (bf16_t*)alloc((size_t)ROWS * D_MODEL * 2);
    bf16_t* Vtb  = (bf16_t*)alloc((size_t)NHB * HD * SEQ * 2);
    bf16_t* Attn = (bf16_t*)alloc((size_t)ROWS * D_MODEL * 2);

    auto cvt = [&](const float* s, bf16_t* d, int n) {
        f32_to_bf16_kernel<<<(n + 255) / 256, 256, 0, stream>>>(s, d, n);
    };
    cvt(x,   xb,  ROWS * D_MODEL);
    cvt(y,   yb,  ROWS * D_MODEL);
    cvt(q_w, wqb, D_MODEL * D_MODEL);
    cvt(k_w, wkb, D_MODEL * D_MODEL);
    cvt(v_w, wvb, D_MODEL * D_MODEL);
    cvt(o_w, wob, D_MODEL * D_MODEL);

    // QKV projections (z selects Q / K / V)
    dim3 gq(D_MODEL / 64, ROWS / 64, 3);
    gemm_qkv_kernel<<<gq, 128, 0, stream>>>(xb, yb, wqb, wkb, wvb,
                                            q_b, k_b, v_b, Qb, Kb, Vb);

    transpose_v_kernel<<<(NHB * HD * SEQ + 255) / 256, 256, 0, stream>>>(Vb, Vtb);

    dim3 ga(SEQ / 16, NHB);
    attn_kernel<<<ga, 128, 0, stream>>>(Qb, Kb, Vtb, Attn);

    dim3 go(D_MODEL / 64, ROWS / 64);
    gemm_out_kernel<<<go, 128, 0, stream>>>(Attn, wob, o_b, out);
}
